// DynamicGraphPredictor_44324062495052
// MI455X (gfx1250) — matrix operand order repaired and checked by reading, not compiled, not run on hardware
//
#include <hip/hip_runtime.h>
#include <hip/hip_bf16.h>
#include <math.h>
#include <stdint.h>

#define BB 4
#define TT 12
#define NN 768
#define FF 32
#define HH 64
#define KT 3
#define LL 2
#define EE 24576
#define BT (BB*TT)            // 48
#define MROWS (BT*NN)         // 36864
#define EPSF 1e-5f

typedef __attribute__((ext_vector_type(16))) _Float16 v16h;
typedef __attribute__((ext_vector_type(8)))  _Float16 v8h;
typedef __attribute__((ext_vector_type(8)))  float    v8f;
typedef __attribute__((ext_vector_type(4)))  unsigned int v4u;
typedef __attribute__((ext_vector_type(8)))  int v8i;
typedef __attribute__((ext_vector_type(4)))  int v4i;

// ---------------- WMMA helpers (CDNA5 16x16x32 f16 -> f32) ----------------

__device__ __forceinline__ v8f zero_v8f() {
  v8f a;
#pragma unroll
  for (int j = 0; j < 8; ++j) a[j] = 0.f;
  return a;
}

__device__ __forceinline__ v16h frag_zero() {
  v16h a;
#pragma unroll
  for (int j = 0; j < 16; ++j) a[j] = (_Float16)0.f;
  return a;
}

// Build 16-bit A fragment (16x32, M x K) from f32 row data.
// lane m=lane%16 holds row M=m; half=lane/16: elements 0..15 -> K=k0+half*16+j.
// p = row base + k0 + half*8; loads start at p + half*8 (32B aligned).
__device__ __forceinline__ v16h frag_a_from_f32(const float* p) {
  p += ((threadIdx.x >> 4) & 1) * 8;
  v8f x1 = *(const v8f*)(p);
  v8f x2 = *(const v8f*)(p + 8);
  v16h a;
#pragma unroll
  for (int j = 0; j < 8; ++j) {
    a[j]     = (_Float16)x1[j];
    a[8 + j] = (_Float16)x2[j];
  }
  return a;
}

// B fragment (32x16, K x N) from [out][K]-transposed weights in global memory.
__device__ __forceinline__ v16h frag_b(const _Float16* __restrict__ wt, int Kdim,
                                       int nt, int k0, int lane) {
  const _Float16* p = wt + (size_t)(nt * 16 + (lane & 15)) * Kdim + k0 + ((lane >> 4) * 16);
  return *(const v16h*)p;
}

// B fragment from LDS with arbitrary (16B-aligned) row stride in f16 elements.
__device__ __forceinline__ v16h frag_b_lds(const _Float16* base, int stride,
                                           int nt, int k0, int lane) {
  const _Float16* p = base + (nt * 16 + (lane & 15)) * stride + k0 + ((lane >> 4) * 16);
  v8h lo = *(const v8h*)p;
  v8h hi = *(const v8h*)(p + 8);
  v16h r;
#pragma unroll
  for (int j = 0; j < 8; ++j) { r[j] = lo[j]; r[8 + j] = hi[j]; }
  return r;
}

#define WMMA(acc, a, b) \
  acc = __builtin_amdgcn_wmma_f32_16x16x32_f16(false, (a), false, (b), (short)0, (acc), false, false)

// C/D layout: element e of acc -> row = rowtile*16 + e + 8*(lane>>4), col = nt*16 + (lane&15)

// ---------------- graph preprocessing ----------------

__global__ void k_deg(const int* __restrict__ rowi, const float* __restrict__ ew,
                      float* __restrict__ deg) {
  int e = blockIdx.x * 256 + threadIdx.x;
  if (e < EE) atomicAdd(&deg[rowi[e]], ew[e]);
}

__global__ void k_dinv(const float* __restrict__ deg, float* __restrict__ dinv) {
  int n = blockIdx.x * 256 + threadIdx.x;
  if (n < NN) {
    float d = deg[n];
    dinv[n] = d > 0.f ? rsqrtf(fmaxf(d, 1e-12f)) : 0.f;
  }
}

__global__ void k_lapw(const int* __restrict__ rowi, const int* __restrict__ coli,
                       const float* __restrict__ ew, const float* __restrict__ dinv,
                       float* __restrict__ lw) {
  int e = blockIdx.x * 256 + threadIdx.x;
  if (e < EE) lw[e] = -dinv[rowi[e]] * ew[e] * dinv[coli[e]];
}

// ---------------- weight preparation (f32 -> f16, [out][K] transposed) ----------------

__global__ void k_prep_inproj(const float* __restrict__ W, _Float16* __restrict__ Wt) {
  int tid = blockIdx.x * 256 + threadIdx.x;               // 64*32
  if (tid < HH * FF) {
    int o = tid / FF, kin = tid % FF;
    Wt[o * FF + kin] = (_Float16)W[kin * HH + o];         // W is (F,H)
  }
}

// tc_w (L,2,3,H,H,KT) -> per (l,dir): Wt[(j*64+o)][kt*64+i] = tc_w[l,dir,j,o,i,kt]
__global__ void k_prep_tconv(const float* __restrict__ W, _Float16* __restrict__ Wt) {
  int tid = blockIdx.x * 256 + threadIdx.x;               // 4*192*192 = 147456
  if (tid < LL * 2 * 192 * 192) {
    int ld  = tid / (192 * 192);
    int rem = tid % (192 * 192);
    int out = rem / 192, k = rem % 192;
    int j = out / HH, o = out % HH;
    int kt = k / HH, i = k % HH;
    float v = W[(((size_t)(ld * 3 + j) * HH + o) * HH + i) * KT + kt];
    Wt[(size_t)ld * 192 * 192 + (size_t)out * 192 + k] = (_Float16)v;
  }
}

// cheb_w (L,K,H,H) -> per l: Wt[d][kk*64+c] = cheb_w[l,kk,c,d]
__global__ void k_prep_cheb(const float* __restrict__ W, _Float16* __restrict__ Wt) {
  int tid = blockIdx.x * 256 + threadIdx.x;               // 2*64*128 = 16384
  if (tid < LL * HH * 128) {
    int l = tid / (HH * 128);
    int rem = tid % (HH * 128);
    int d = rem / 128, k = rem % 128;
    int kk = k / HH, c = k % HH;
    float v = W[(((size_t)(l * 2 + kk) * HH) + c) * HH + d];
    Wt[(size_t)l * HH * 128 + (size_t)d * 128 + k] = (_Float16)v;
  }
}

// out1_w (2H,H/2) -> Wt[out][kin]: out<32 -> a-proj, out>=32 -> c-proj
__global__ void k_prep_head(const float* __restrict__ W, _Float16* __restrict__ Wt) {
  int tid = blockIdx.x * 256 + threadIdx.x;               // 64*64
  if (tid < 64 * 64) {
    int out = tid / 64, kin = tid % 64;
    float v = (out < 32) ? W[kin * 32 + out] : W[(64 + kin) * 32 + (out - 32)];
    Wt[out * 64 + kin] = (_Float16)v;
  }
}

// ---------------- GEMMs ----------------

// H = X @ in_proj_w + b ;  M=36864, K=32, Nout=64.  block=128 (4 waves), grid=576
__global__ void k_gemm_inproj(const float* __restrict__ X, const _Float16* __restrict__ Wt,
                              const float* __restrict__ bias, float* __restrict__ Hh) {
  int lane = threadIdx.x & 31, wid = threadIdx.x >> 5;
  int rt = blockIdx.x * 4 + wid;
  int half = lane >> 4, m = lane & 15;
  int r0 = rt * 16;
  v8f acc[4];
#pragma unroll
  for (int q = 0; q < 4; ++q) acc[q] = zero_v8f();
  v16h a = frag_a_from_f32(X + (size_t)(r0 + m) * FF + half * 8);
#pragma unroll
  for (int q = 0; q < 4; ++q) {
    v16h b = frag_b(Wt, FF, q, 0, lane);
    WMMA(acc[q], a, b);
  }
  int jc = lane & 15;
#pragma unroll
  for (int q = 0; q < 4; ++q) {
    int j = q * 16 + jc;
    float bj = bias[j];
#pragma unroll
    for (int e = 0; e < 8; ++e)
      Hh[(size_t)(r0 + e + 8 * half) * HH + j] = acc[q][e] + bj;
  }
}

// Gated temporal conv as one K=192 GEMM with fused time-gather + GLU epilogue.
// B (192x192 f16) staged into LDS in two K-phases of 96 via async global->LDS
// copies (ASYNCcnt path); padded stride 104 f16 (52 dwords) is bank-conflict-free.
#define TCS 104
__global__ void __launch_bounds__(128) k_gemm_tconv(
    const float* __restrict__ X, const _Float16* __restrict__ Wt,
    const float* __restrict__ bias3, float* __restrict__ Z) {
  __shared__ _Float16 sB[192 * TCS];    // 39936 B
  int tid = threadIdx.x;
  int lane = threadIdx.x & 31, wid = threadIdx.x >> 5;
  int rt = blockIdx.x * 4 + wid;
  int half = lane >> 4, m = lane & 15;
  int r0 = rt * 16;
  int bt = r0 / NN;          // uniform across tile (16 | 768)
  int t = bt % TT;
  v8f acc[12];
#pragma unroll
  for (int c = 0; c < 12; ++c) acc[c] = zero_v8f();

  for (int phase = 0; phase < 2; ++phase) {
    // async-stage K in [phase*96, phase*96+96): 192 rows x 12 b128 chunks
    for (int c = tid; c < 192 * 12; c += 128) {
      int row = c / 12, cw = c % 12;
      const _Float16* gp = Wt + (size_t)row * 192 + phase * 96 + cw * 8;
      unsigned lo = (unsigned)(uintptr_t)&sB[row * TCS + cw * 8];
      asm volatile("global_load_async_to_lds_b128 %0, %1, off"
                   :: "v"(lo), "v"(gp) : "memory");
    }
    asm volatile("s_wait_asynccnt 0x0" ::: "memory");
    __syncthreads();

#pragma unroll
    for (int ks2 = 0; ks2 < 3; ++ks2) {
      int ks = phase * 3 + ks2;
      int kt = ks >> 1;
      int tsrc = t + kt - 1;
      v16h a;
      if (tsrc >= 0 && tsrc < TT) {      // wave-uniform branch, EXEC stays all-1
        const float* p = X + ((long)(r0 + m) + (long)(kt - 1) * NN) * HH
                           + ((ks & 1) * 32) + half * 8;
        a = frag_a_from_f32(p);
      } else {
        a = frag_zero();
      }
#pragma unroll
      for (int nt = 0; nt < 12; ++nt) {
        v16h b = frag_b_lds(sB, TCS, nt, ks2 * 32, lane);
        WMMA(acc[nt], a, b);
      }
    }
    __syncthreads();   // staged data fully consumed before next phase overwrites
  }

  int jc = lane & 15;
#pragma unroll
  for (int q = 0; q < 4; ++q) {
    int j = q * 16 + jc;
    float b0 = bias3[j], b1 = bias3[64 + j], b2 = bias3[128 + j];
#pragma unroll
    for (int e = 0; e < 8; ++e) {
      float c0 = acc[q][e] + b0;
      float c1 = acc[4 + q][e] + b1;
      float c2 = acc[8 + q][e] + b2;
      float g = c0 * (1.f / (1.f + __expf(-c1))) + c2;
      Z[(size_t)(r0 + e + 8 * half) * HH + j] = g > 0.f ? g : 0.f;
    }
  }
}

// Chebyshev: Z2 = relu([Z1 | AGG] @ [theta0;theta1] + bias) ; K=128, Nout=64.
// B (64x128 f16 = 16KB contiguous) staged into LDS by the Tensor Data Mover:
// one wave issues tensor_load_to_lds with a minimal 2D D# (dim0=tile0=8192
// 2-byte elements, dim1=1), waits TENSORcnt, then the block reads from LDS.
__global__ void __launch_bounds__(128) k_gemm_cheb(
    const float* __restrict__ Z1, const float* __restrict__ AGG,
    const _Float16* __restrict__ Wt, const float* __restrict__ bias,
    float* __restrict__ Z2) {
  __shared__ _Float16 sC[64 * 128];     // 16384 B
  int lane = threadIdx.x & 31, wid = threadIdx.x >> 5;
  if (wid == 0) {
    unsigned long long ga = (unsigned long long)(uintptr_t)Wt;
    unsigned ldso = (unsigned)(uintptr_t)&sC[0];
    const unsigned NEL = 64u * 128u;    // 8192 elements of 2 bytes
    v4u g0;
    g0[0] = 1u;                                         // count=1, user D#
    g0[1] = ldso;                                       // lds_addr (bytes)
    g0[2] = (unsigned)(ga & 0xffffffffu);               // global_addr[31:0]
    g0[3] = (unsigned)((ga >> 32) & 0x1ffffffu) | (2u << 30);  // addr[56:32] | type=2
    v8i g1;
    g1[0] = (int)(1u << 16);                 // workgroup_mask=0, data_size=1 (2B)
    g1[1] = (int)((NEL & 0xffffu) << 16);    // tensor_dim0[15:0]
    g1[2] = (int)(((NEL >> 16) & 0xffffu) | (1u << 16)); // tensor_dim0[31:16] | tensor_dim1=1
    g1[3] = (int)((NEL & 0xffffu) << 16);    // tensor_dim1[31:16]=0 | tile_dim0=8192
    g1[4] = 1;                               // tile_dim1=1, tile_dim2=0
    g1[5] = (int)NEL;                        // tensor_dim0_stride[31:0]
    g1[6] = 0;                               // stride0[47:32]=0, stride1[15:0]=0
    g1[7] = 0;                               // stride1[47:16]=0
    v4i g2; g2[0] = 0; g2[1] = 0; g2[2] = 0; g2[3] = 0;
    v4i g3; g3[0] = 0; g3[1] = 0; g3[2] = 0; g3[3] = 0;
    v8i g4;
#pragma unroll
    for (int j = 0; j < 8; ++j) g4[j] = 0;
    __builtin_amdgcn_tensor_load_to_lds(g0, g1, g2, g3, g4, 0);
    __builtin_amdgcn_s_wait_tensorcnt(0);
  }
  __syncthreads();

  int rt = blockIdx.x * 4 + wid;
  int half = lane >> 4, m = lane & 15;
  int r0 = rt * 16;
  v8f acc[4];
#pragma unroll
  for (int q = 0; q < 4; ++q) acc[q] = zero_v8f();
#pragma unroll
  for (int ks = 0; ks < 4; ++ks) {
    const float* src = (ks < 2) ? Z1 : AGG;
    const float* p = src + (size_t)(r0 + m) * HH + ((ks & 1) * 32) + half * 8;
    v16h a = frag_a_from_f32(p);
#pragma unroll
    for (int nt = 0; nt < 4; ++nt) {
      v16h b = frag_b_lds(sC, 128, nt, ks * 32, lane);
      WMMA(acc[nt], a, b);
    }
  }
  int jc = lane & 15;
#pragma unroll
  for (int q = 0; q < 4; ++q) {
    int j = q * 16 + jc;
    float bj = bias[j];
#pragma unroll
    for (int e = 0; e < 8; ++e) {
      float v = acc[q][e] + bj;
      Z2[(size_t)(r0 + e + 8 * half) * HH + j] = v > 0.f ? v : 0.f;
    }
  }
}

// Head GEMM: AC[b*N+n][0:32]=emb@W_a, [32:64]=emb@W_c+b ; emb = h[:, T-1]
__global__ void k_gemm_head(const float* __restrict__ Hh, const _Float16* __restrict__ Wt,
                            const float* __restrict__ b1, float* __restrict__ AC) {
  int lane = threadIdx.x & 31, wid = threadIdx.x >> 5;
  int rt = blockIdx.x * 4 + wid;                // 192 tiles, grid=48
  int half = lane >> 4, m = lane & 15;
  int r0 = rt * 16;
  int re = r0 + m;
  int bidx = re / NN, n = re % NN;
  const float* hrow = Hh + ((size_t)(bidx * TT + (TT - 1)) * NN + n) * HH;
  v8f acc[4];
#pragma unroll
  for (int q = 0; q < 4; ++q) acc[q] = zero_v8f();
#pragma unroll
  for (int ks = 0; ks < 2; ++ks) {
    v16h a = frag_a_from_f32(hrow + ks * 32 + half * 8);
#pragma unroll
    for (int nt = 0; nt < 4; ++nt) {
      v16h b = frag_b(Wt, 64, nt, ks * 32, lane);
      WMMA(acc[nt], a, b);
    }
  }
  int jc = lane & 15;
#pragma unroll
  for (int q = 0; q < 4; ++q) {
    int j = q * 16 + jc;
    float bj = (j >= 32) ? b1[j - 32] : 0.f;
#pragma unroll
    for (int e = 0; e < 8; ++e)
      AC[(size_t)(r0 + e + 8 * half) * 64 + j] = acc[q][e] + bj;
  }
}

// ---------------- sparse aggregation (L-hat message passing) ----------------

__global__ void k_scatter(const float* __restrict__ Z, const int* __restrict__ rowi,
                          const int* __restrict__ coli, const float* __restrict__ lw,
                          float* __restrict__ AGG) {
  int tid = blockIdx.x * 256 + threadIdx.x;     // E*BT*16 = 18,874,368
  int e = tid / (BT * 16);
  int rem = tid % (BT * 16);
  int bt = rem >> 4, hq = rem & 15;
  float w = lw[e];
  int cs = coli[e], rs = rowi[e];
  float4 v = *(const float4*)(Z + ((size_t)bt * NN + cs) * HH + hq * 4);
  float* dst = AGG + ((size_t)bt * NN + rs) * HH + hq * 4;
  atomicAdd(dst + 0, v.x * w);
  atomicAdd(dst + 1, v.y * w);
  atomicAdd(dst + 2, v.z * w);
  atomicAdd(dst + 3, v.w * w);
}

// ---------------- BN stats + fused BN/LN/residual ----------------

__global__ void k_bnstats(const float* __restrict__ Z, float* __restrict__ stat) {
  __shared__ float s1[256], s2[256];
  int n = blockIdx.x, tid = threadIdx.x;
  float s = 0.f, sq = 0.f;
  for (int i = tid; i < BT * HH; i += 256) {
    int bt = i >> 6, hh = i & 63;
    float v = Z[((size_t)bt * NN + n) * HH + hh];
    s += v; sq += v * v;
  }
  s1[tid] = s; s2[tid] = sq;
  __syncthreads();
  for (int off = 128; off > 0; off >>= 1) {
    if (tid < off) { s1[tid] += s1[tid + off]; s2[tid] += s2[tid + off]; }
    __syncthreads();
  }
  if (tid == 0) {
    float mean = s1[0] / (float)(BT * HH);
    float var = s2[0] / (float)(BT * HH) - mean * mean;
    stat[2 * n] = mean;
    stat[2 * n + 1] = rsqrtf(var + EPSF);
  }
}

__global__ void k_bnln_res(const float* __restrict__ Z, const float* __restrict__ stat,
                           const float* __restrict__ bng, const float* __restrict__ bnb,
                           const float* __restrict__ lng, const float* __restrict__ lnb,
                           float* __restrict__ Hh) {
  int lane = threadIdx.x & 31, wid = threadIdx.x >> 5;
  int r = blockIdx.x * 8 + wid;                 // 36864 rows, grid=4608
  int n = r % NN;
  float mu = stat[2 * n], ri = stat[2 * n + 1];
  float g = bng[n], bb = bnb[n];
  int h0 = lane, h1 = lane + 32;
  float v0 = (Z[(size_t)r * HH + h0] - mu) * ri * g + bb;
  float v1 = (Z[(size_t)r * HH + h1] - mu) * ri * g + bb;
  float s = v0 + v1, sq = v0 * v0 + v1 * v1;
#pragma unroll
  for (int o = 16; o > 0; o >>= 1) {
    s  += __shfl_xor(s, o, 32);
    sq += __shfl_xor(sq, o, 32);
  }
  float mean = s / 64.f;
  float var = sq / 64.f - mean * mean;
  float rv = rsqrtf(var + EPSF);
  Hh[(size_t)r * HH + h0] += (v0 - mean) * rv * lng[h0] + lnb[h0];
  Hh[(size_t)r * HH + h1] += (v1 - mean) * rv * lng[h1] + lnb[h1];
}

// ---------------- head final: relu-add -> LN(32) -> dot -> sigmoid ----------------

__global__ void k_head_final(const float* __restrict__ AC, const float* __restrict__ olng,
                             const float* __restrict__ olnb, const float* __restrict__ w2,
                             const float* __restrict__ b2, float* __restrict__ out) {
  int tid = blockIdx.x * 256 + threadIdx.x;     // B*N*N = 2,359,296
  int b = tid / (NN * NN);
  int rem = tid % (NN * NN);
  int i = rem / NN, j = rem % NN;
  const float* ap = AC + ((size_t)b * NN + i) * 64;
  const float* cp = AC + ((size_t)b * NN + j) * 64 + 32;
  float p[32];
  float s = 0.f, sq = 0.f;
#pragma unroll
  for (int h = 0; h < 32; ++h) {
    float v = ap[h] + cp[h];
    v = v > 0.f ? v : 0.f;
    p[h] = v; s += v; sq += v * v;
  }
  float mean = s / 32.f;
  float var = sq / 32.f - mean * mean;
  float rv = rsqrtf(var + EPSF);
  float acc = 0.f;
#pragma unroll
  for (int h = 0; h < 32; ++h)
    acc += ((p[h] - mean) * rv * olng[h] + olnb[h]) * w2[h];
  float logit = acc + b2[0];
  out[tid] = 1.f / (1.f + __expf(-logit));
}

// ---------------- launch ----------------

static inline size_t align256(size_t x) { return (x + 255) & ~(size_t)255; }

extern "C" void kernel_launch(void* const* d_in, const int* in_sizes, int n_in,
                              void* d_out, int out_size, void* d_ws, size_t ws_size,
                              hipStream_t stream) {
  const float* x        = (const float*)d_in[0];
  const int*   ei       = (const int*)d_in[1];
  const float* ew       = (const float*)d_in[2];
  const float* in_w     = (const float*)d_in[3];
  const float* in_b     = (const float*)d_in[4];
  const float* tc_w     = (const float*)d_in[5];
  const float* tc_b     = (const float*)d_in[6];
  const float* cheb_w   = (const float*)d_in[7];
  const float* cheb_b   = (const float*)d_in[8];
  const float* bn_g     = (const float*)d_in[9];
  const float* bn_b     = (const float*)d_in[10];
  const float* ln_g     = (const float*)d_in[11];
  const float* ln_b     = (const float*)d_in[12];
  const float* out1_w   = (const float*)d_in[13];
  const float* out1_b   = (const float*)d_in[14];
  const float* oln_g    = (const float*)d_in[15];
  const float* oln_b    = (const float*)d_in[16];
  const float* out2_w   = (const float*)d_in[17];
  const float* out2_b   = (const float*)d_in[18];
  const int* rowi = ei;
  const int* coli = ei + EE;

  char* ws = (char*)d_ws;
  size_t o = 0;
  float*    deg  = (float*)(ws + o);    o = align256(o + NN * 4);
  float*    dinv = (float*)(ws + o);    o = align256(o + NN * 4);
  float*    lw   = (float*)(ws + o);    o = align256(o + EE * 4);
  _Float16* WtIn = (_Float16*)(ws + o); o = align256(o + HH * FF * 2);
  _Float16* WtTc = (_Float16*)(ws + o); o = align256(o + (size_t)LL * 2 * 192 * 192 * 2);
  _Float16* WtCh = (_Float16*)(ws + o); o = align256(o + (size_t)LL * HH * 128 * 2);
  _Float16* WtHd = (_Float16*)(ws + o); o = align256(o + 64 * 64 * 2);
  float*    Hh   = (float*)(ws + o);    o = align256(o + (size_t)MROWS * HH * 4);
  float*    Z1   = (float*)(ws + o);    o = align256(o + (size_t)MROWS * HH * 4);
  float*    Z2   = (float*)(ws + o);    o = align256(o + (size_t)MROWS * HH * 4);
  float*    AGG  = (float*)(ws + o);    o = align256(o + (size_t)MROWS * HH * 4);
  float*    STAT = (float*)(ws + o);    o = align256(o + NN * 2 * 4);
  float*    AC   = (float*)(ws + o);    o = align256(o + (size_t)BB * NN * 64 * 4);
  (void)ws_size; (void)in_sizes; (void)n_in; (void)out_size;

  // graph normalization
  (void)hipMemsetAsync(deg, 0, NN * 4, stream);
  k_deg <<<(EE + 255) / 256, 256, 0, stream>>>(rowi, ew, deg);
  k_dinv<<<(NN + 255) / 256, 256, 0, stream>>>(deg, dinv);
  k_lapw<<<(EE + 255) / 256, 256, 0, stream>>>(rowi, coli, ew, dinv, lw);

  // f16 transposed weight prep
  k_prep_inproj<<<(HH * FF + 255) / 256, 256, 0, stream>>>(in_w, WtIn);
  k_prep_tconv <<<(LL * 2 * 192 * 192 + 255) / 256, 256, 0, stream>>>(tc_w, WtTc);
  k_prep_cheb  <<<(LL * HH * 128 + 255) / 256, 256, 0, stream>>>(cheb_w, WtCh);
  k_prep_head  <<<(64 * 64 + 255) / 256, 256, 0, stream>>>(out1_w, WtHd);

  // input projection
  k_gemm_inproj<<<MROWS / 16 / 4, 128, 0, stream>>>(x, WtIn, in_b, Hh);

  for (int l = 0; l < LL; ++l) {
    // tconv #1 (gated GLU fused, async-LDS-staged B)
    k_gemm_tconv<<<MROWS / 16 / 4, 128, 0, stream>>>(
        Hh, WtTc + (size_t)(l * 2 + 0) * 192 * 192, tc_b + (size_t)(l * 2 + 0) * 3 * HH, Z1);
    // sparse aggregation
    (void)hipMemsetAsync(AGG, 0, (size_t)MROWS * HH * 4, stream);
    k_scatter<<<(EE * BT * 16) / 256, 256, 0, stream>>>(Z1, rowi, coli, lw, AGG);
    // chebyshev (bias + relu fused, TDM-staged B)
    k_gemm_cheb<<<MROWS / 16 / 4, 128, 0, stream>>>(
        Z1, AGG, WtCh + (size_t)l * HH * 128, cheb_b + (size_t)l * HH, Z2);
    // tconv #2
    k_gemm_tconv<<<MROWS / 16 / 4, 128, 0, stream>>>(
        Z2, WtTc + (size_t)(l * 2 + 1) * 192 * 192, tc_b + (size_t)(l * 2 + 1) * 3 * HH, Z1);
    // BN stats + fused BN/LN/residual
    k_bnstats<<<NN, 256, 0, stream>>>(Z1, STAT);
    k_bnln_res<<<MROWS / 8, 256, 0, stream>>>(
        Z1, STAT, bn_g + (size_t)l * NN, bn_b + (size_t)l * NN,
        ln_g + (size_t)l * HH, ln_b + (size_t)l * HH, Hh);
  }

  // head
  k_gemm_head<<<(BB * NN) / 16 / 4, 128, 0, stream>>>(Hh, WtHd, out1_b, AC);
  k_head_final<<<(BB * NN * NN) / 256, 256, 0, stream>>>(
      AC, oln_g, oln_b, out2_w, out2_b, (float*)d_out);
}
